// StochasticDALSTM_70231305224592
// MI455X (gfx1250) — matrix operand order repaired and checked by reading, not compiled, not run on hardware
//
#include <hip/hip_runtime.h>
#include <hip/hip_bf16.h>

// ---------------------------------------------------------------------------
// StochasticDALSTM for MI455X (gfx1250, wave32, WMMA bf16)
//
// Pipeline per timestep t (all on `stream`):
//   prep1 : A1[g,b,0:64]      = bf16( x[b,t,:] * zx1[g,b,:] )
//   gemm  : pre1[g,b,n]       = sum_k A1[g,b,k] * Wcat1[g,n,k]      (K=1088)
//   elem1 : LSTM cell 1 -> h1;  A1[g,b,64+h] = bf16(h1*zh1),
//                               A2[g,b,h]    = bf16(h1*zx2)
//   gemm  : pre2[g,b,n]       = sum_k A2[g,b,k] * Wcat2[g,n,k]      (K=2048)
//   elem2 : LSTM cell 2 -> h2;  A2[g,b,1024+h] = bf16(h2*zh2), h2last = h2
// Tail: BN stats over h2last (only t=127 matters), masked linear head.
//
// GEMM tiling: each wave owns a 32(M) x 32(N) register tile (4 WMMA
// accumulators, 2 A-frags + 2 B-frags per k-step = 2 b128 loads per WMMA);
// 2 waves per block split N -> 1024 waves per GEMM for occupancy.
// Weights (26 MB bf16) stay L2-resident across all 128 steps.
// ---------------------------------------------------------------------------

#define C_EPS    1e-7f
#define C_BNEPS  1e-5f
#define C_INPUT  64
#define C_HID    1024
#define C_T      128
#define C_B      256

typedef __attribute__((ext_vector_type(16))) __bf16 v16bf;
typedef __attribute__((ext_vector_type(8)))  float  v8f;

union FragU { v16bf v; uint4 q[2]; };

// A-fragment (16x32, 16-bit): lane = M row; lanes 0-15 hold K=0..15,
// lanes 16-31 hold K=16..31 (contiguous, matching load_fragB).
__device__ __forceinline__ v16bf load_fragA(const __bf16* __restrict__ base,
                                            int ld, int k0, int lane) {
  int r = lane & 15;
  int sel = lane >> 4;
  const __bf16* p = base + (size_t)r * ld + k0 + sel * 16;
  FragU f;
  f.q[0] = *reinterpret_cast<const uint4*>(p);
  f.q[1] = *reinterpret_cast<const uint4*>(p + 8);
  return f.v;
}

// B-fragment (32x16, 16-bit): lane = N col; lanes 0-15 hold K=0..15,
// lanes 16-31 hold K=16..31 (contiguous), matching load_fragA.
__device__ __forceinline__ v16bf load_fragB(const __bf16* __restrict__ base,
                                            int ld, int k0, int lane) {
  int n = lane & 15;
  int sel = lane >> 4;
  const __bf16* p = base + (size_t)n * ld + k0 + sel * 16;
  FragU f;
  f.q[0] = *reinterpret_cast<const uint4*>(p);
  f.q[1] = *reinterpret_cast<const uint4*>(p + 8);
  return f.v;
}

// C = A * W^T per gate. A: [4][256][K] bf16, W: [4][1024][K] bf16,
// C: [4][256][1024] f32. Wave computes 32(M) x 32(N); 2 waves/block over N.
__global__ __launch_bounds__(64)
void gemm_bf16_kernel(const __bf16* __restrict__ A,
                      const __bf16* __restrict__ W,
                      float* __restrict__ C, int K) {
  const int M = C_B, N = C_HID;
  int lane = threadIdx.x & 31;
  int wid  = threadIdx.x >> 5;                 // 0..1: N sub-tile
  int n0 = blockIdx.x * 64 + wid * 32;
  int m0 = blockIdx.y * 32;
  int g  = blockIdx.z;

  const __bf16* A0 = A + (size_t)g * M * K + (size_t)m0 * K;
  const __bf16* A1 = A0 + (size_t)16 * K;
  const __bf16* W0 = W + (size_t)g * N * K + (size_t)n0 * K;
  const __bf16* W1 = W0 + (size_t)16 * K;

  v8f c00 = {}; v8f c01 = {}; v8f c10 = {}; v8f c11 = {};
  for (int k0 = 0; k0 < K; k0 += 32) {
    v16bf a0 = load_fragA(A0, K, k0, lane);
    v16bf a1 = load_fragA(A1, K, k0, lane);
    v16bf b0 = load_fragB(W0, K, k0, lane);
    v16bf b1 = load_fragB(W1, K, k0, lane);
    c00 = __builtin_amdgcn_wmma_f32_16x16x32_bf16(false, a0, false, b0,
                                                  (short)0, c00, false, false);
    c01 = __builtin_amdgcn_wmma_f32_16x16x32_bf16(false, a0, false, b1,
                                                  (short)0, c01, false, false);
    c10 = __builtin_amdgcn_wmma_f32_16x16x32_bf16(false, a1, false, b0,
                                                  (short)0, c10, false, false);
    c11 = __builtin_amdgcn_wmma_f32_16x16x32_bf16(false, a1, false, b1,
                                                  (short)0, c11, false, false);
  }

  // C/D layout: VGPR v, lane L -> m = v + 8*(L>>4), n = L&15.
  float* Cg = C + (size_t)g * M * N;
  int n = lane & 15;
  int sel = lane >> 4;
#pragma unroll
  for (int v = 0; v < 8; ++v) {
    int mA = m0 + v + 8 * sel;
    int mB = mA + 16;
    Cg[(size_t)mA * N + n0 + n]      = c00[v];
    Cg[(size_t)mA * N + n0 + 16 + n] = c01[v];
    Cg[(size_t)mB * N + n0 + n]      = c10[v];
    Cg[(size_t)mB * N + n0 + 16 + n] = c11[v];
  }
}

// concrete_mask: z = (1 - sigmoid((logit(p) + logit(u)) / 0.1)) / (1 - p)
__global__ void mask_kernel(const float* __restrict__ p_logit,
                            const float* __restrict__ u,
                            float* __restrict__ z, int n) {
  int i = blockIdx.x * blockDim.x + threadIdx.x;
  if (i >= n) return;
  float p  = 1.f / (1.f + __expf(-p_logit[0]));
  float uu = u[i];
  float t = (__logf(p + C_EPS) - __logf(1.f - p + C_EPS) +
             __logf(uu + C_EPS) - __logf(1.f - uu + C_EPS)) * 10.0f;
  float zz = 1.f - 1.f / (1.f + __expf(-t));
  z[i] = zz / (1.f - p);
}

// Wcat[g][n][0:Kx] = Wx[g][n][:], Wcat[g][n][Kx:Kx+Kh] = Wh[g][n][:]  (bf16)
__global__ void wcat_kernel(const float* __restrict__ Wx,
                            const float* __restrict__ Wh,
                            __hip_bfloat16* __restrict__ Wcat,
                            int Kx, int Kh) {
  long K = Kx + Kh;
  long total = 4L * C_HID * K;
  for (long i = (long)blockIdx.x * blockDim.x + threadIdx.x; i < total;
       i += (long)gridDim.x * blockDim.x) {
    long gn = i / K;
    int  k  = (int)(i % K);
    float v = (k < Kx) ? Wx[gn * Kx + k] : Wh[gn * Kh + (k - Kx)];
    Wcat[i] = __hip_bfloat16(v);
  }
}

__global__ void zero_kernel(unsigned int* __restrict__ p, long n32) {
  for (long i = (long)blockIdx.x * blockDim.x + threadIdx.x; i < n32;
       i += (long)gridDim.x * blockDim.x)
    p[i] = 0u;
}

// A1[g,b,0:64] = bf16(x[b,t,:] * zx1[g,b,:])
__global__ void prep1_kernel(const float* __restrict__ x,
                             const float* __restrict__ zx1,
                             __hip_bfloat16* __restrict__ A1, int t) {
  int idx = blockIdx.x * blockDim.x + threadIdx.x;  // g*B*64 + b*64 + i
  if (idx >= 4 * C_B * C_INPUT) return;
  int g = idx / (C_B * C_INPUT);
  int r = idx - g * (C_B * C_INPUT);
  int b = r / C_INPUT;
  int i = r - b * C_INPUT;
  float v = x[((size_t)b * C_T + t) * C_INPUT + i] * zx1[idx];
  A1[(size_t)g * C_B * 1088 + (size_t)b * 1088 + i] = __hip_bfloat16(v);
}

__device__ __forceinline__ float sigf(float x) {
  return 1.f / (1.f + __expf(-x));
}

// LSTM cell layer 1: consumes pre1, updates c1, emits masked bf16 operands.
__global__ void elem1_kernel(const float* __restrict__ pre,
                             const float* __restrict__ bx,
                             const float* __restrict__ bh,
                             const float* __restrict__ zh1,
                             const float* __restrict__ zx2,
                             float* __restrict__ cst,
                             __hip_bfloat16* __restrict__ A1,
                             __hip_bfloat16* __restrict__ A2) {
  int idx = blockIdx.x * blockDim.x + threadIdx.x;  // b*H + h
  if (idx >= C_B * C_HID) return;
  int b = idx >> 10, h = idx & (C_HID - 1);
  const int GH = C_B * C_HID;
  float pi = pre[0 * GH + idx] + bx[h]           + bh[h];
  float pf = pre[1 * GH + idx] + bx[C_HID + h]   + bh[C_HID + h];
  float po = pre[2 * GH + idx] + bx[2*C_HID + h] + bh[2*C_HID + h];
  float pg = pre[3 * GH + idx] + bx[3*C_HID + h] + bh[3*C_HID + h];
  float ig = sigf(pi), fg = sigf(pf), og = sigf(po), gg = tanhf(pg);
  float c = fg * cst[idx] + ig * gg;
  cst[idx] = c;
  float hh = og * tanhf(c);
#pragma unroll
  for (int g = 0; g < 4; ++g) {
    A1[(size_t)g * C_B * 1088 + (size_t)b * 1088 + 64 + h] =
        __hip_bfloat16(hh * zh1[g * GH + idx]);
    A2[(size_t)g * C_B * 2048 + (size_t)b * 2048 + h] =
        __hip_bfloat16(hh * zx2[g * GH + idx]);
  }
}

// LSTM cell layer 2: consumes pre2, updates c2, emits recurrent operand + h2.
__global__ void elem2_kernel(const float* __restrict__ pre,
                             const float* __restrict__ bx,
                             const float* __restrict__ bh,
                             const float* __restrict__ zh2,
                             float* __restrict__ cst,
                             __hip_bfloat16* __restrict__ A2,
                             float* __restrict__ h2last) {
  int idx = blockIdx.x * blockDim.x + threadIdx.x;
  if (idx >= C_B * C_HID) return;
  int b = idx >> 10, h = idx & (C_HID - 1);
  const int GH = C_B * C_HID;
  float pi = pre[0 * GH + idx] + bx[h]           + bh[h];
  float pf = pre[1 * GH + idx] + bx[C_HID + h]   + bh[C_HID + h];
  float po = pre[2 * GH + idx] + bx[2*C_HID + h] + bh[2*C_HID + h];
  float pg = pre[3 * GH + idx] + bx[3*C_HID + h] + bh[3*C_HID + h];
  float ig = sigf(pi), fg = sigf(pf), og = sigf(po), gg = tanhf(pg);
  float c = fg * cst[idx] + ig * gg;
  cst[idx] = c;
  float hh = og * tanhf(c);
#pragma unroll
  for (int g = 0; g < 4; ++g)
    A2[(size_t)g * C_B * 2048 + (size_t)b * 2048 + C_HID + h] =
        __hip_bfloat16(hh * zh2[g * GH + idx]);
  h2last[idx] = hh;
}

// BN stats over h2last (256*1024 elements) -> stats[0]=mean, stats[1]=var
__global__ void stats_kernel(const float* __restrict__ h2,
                             float* __restrict__ stats) {
  __shared__ float ssum[1024];
  __shared__ float ssq[1024];
  float s = 0.f, q = 0.f;
  for (int i = threadIdx.x; i < C_B * C_HID; i += blockDim.x) {
    float v = h2[i];
    s += v; q += v * v;
  }
  ssum[threadIdx.x] = s; ssq[threadIdx.x] = q;
  __syncthreads();
  for (int off = blockDim.x >> 1; off > 0; off >>= 1) {
    if (threadIdx.x < off) {
      ssum[threadIdx.x] += ssum[threadIdx.x + off];
      ssq[threadIdx.x]  += ssq[threadIdx.x + off];
    }
    __syncthreads();
  }
  if (threadIdx.x == 0) {
    const float inv = 1.f / (float)(C_B * C_HID);
    float mean = ssum[0] * inv;
    stats[0] = mean;
    stats[1] = ssq[0] * inv - mean * mean;
  }
}

// mu[b] = sum_h ((norm(h2last)*gamma[127]+beta[127]) * z_out * Wmu) + bmu
__global__ void out_kernel(const float* __restrict__ h2,
                           const float* __restrict__ stats,
                           const float* __restrict__ z_out,
                           const float* __restrict__ Wmu,
                           const float* __restrict__ bmu,
                           const float* __restrict__ gamma,
                           const float* __restrict__ beta,
                           float* __restrict__ out) {
  __shared__ float red[256];
  int b = blockIdx.x;
  float mean = stats[0];
  float rstd = rsqrtf(stats[1] + C_BNEPS);
  float gm = gamma[C_T - 1], bt = beta[C_T - 1];
  float acc = 0.f;
  for (int h = threadIdx.x; h < C_HID; h += blockDim.x) {
    float v = (h2[(size_t)b * C_HID + h] - mean) * rstd * gm + bt;
    acc += v * z_out[(size_t)b * C_HID + h] * Wmu[h];
  }
  red[threadIdx.x] = acc;
  __syncthreads();
  for (int off = blockDim.x >> 1; off > 0; off >>= 1) {
    if (threadIdx.x < off) red[threadIdx.x] += red[threadIdx.x + off];
    __syncthreads();
  }
  if (threadIdx.x == 0) out[b] = red[0] + bmu[0];
}

extern "C" void kernel_launch(void* const* d_in, const int* in_sizes, int n_in,
                              void* d_out, int out_size, void* d_ws, size_t ws_size,
                              hipStream_t stream) {
  (void)in_sizes; (void)n_in; (void)out_size; (void)ws_size;
  const float* x     = (const float*)d_in[0];
  const float* Wx1   = (const float*)d_in[1];
  const float* bx1   = (const float*)d_in[2];
  const float* Wh1   = (const float*)d_in[3];
  const float* bh1   = (const float*)d_in[4];
  const float* p1    = (const float*)d_in[5];
  const float* ux1   = (const float*)d_in[6];
  const float* uh1   = (const float*)d_in[7];
  const float* Wx2   = (const float*)d_in[8];
  const float* bx2   = (const float*)d_in[9];
  const float* Wh2   = (const float*)d_in[10];
  const float* bh2   = (const float*)d_in[11];
  const float* p2    = (const float*)d_in[12];
  const float* ux2   = (const float*)d_in[13];
  const float* uh2   = (const float*)d_in[14];
  const float* Wmu   = (const float*)d_in[15];
  const float* bmu   = (const float*)d_in[16];
  const float* p_out = (const float*)d_in[17];
  const float* u_out = (const float*)d_in[18];
  const float* gamma = (const float*)d_in[19];
  const float* beta  = (const float*)d_in[20];
  float* out = (float*)d_out;

  // -------- workspace layout (≈57 MB, 256B aligned) --------
  char* w = (char*)d_ws;
  auto alloc = [&](size_t bytes) -> void* {
    void* p = (void*)w;
    w += (bytes + 255) & ~(size_t)255;
    return p;
  };
  float* zx1  = (float*)alloc(4L * C_B * C_INPUT * 4);
  float* zh1  = (float*)alloc(4L * C_B * C_HID * 4);
  float* zx2  = (float*)alloc(4L * C_B * C_HID * 4);
  float* zh2  = (float*)alloc(4L * C_B * C_HID * 4);
  float* zo   = (float*)alloc((size_t)C_B * C_HID * 4);
  __hip_bfloat16* Wcat1 = (__hip_bfloat16*)alloc(4L * C_HID * 1088 * 2);
  __hip_bfloat16* Wcat2 = (__hip_bfloat16*)alloc(4L * C_HID * 2048 * 2);
  __hip_bfloat16* A1    = (__hip_bfloat16*)alloc(4L * C_B * 1088 * 2);
  __hip_bfloat16* A2    = (__hip_bfloat16*)alloc(4L * C_B * 2048 * 2);
  float* pre1   = (float*)alloc(4L * C_B * C_HID * 4);
  float* pre2   = (float*)alloc(4L * C_B * C_HID * 4);
  float* c1     = (float*)alloc((size_t)C_B * C_HID * 4);
  float* c2     = (float*)alloc((size_t)C_B * C_HID * 4);
  float* h2last = (float*)alloc((size_t)C_B * C_HID * 4);
  float* stats  = (float*)alloc(256);

  const int GH  = C_B * C_HID;   // 262144
  const int TPB = 256;

  // -------- setup --------
  mask_kernel<<<(4 * C_B * C_INPUT + TPB - 1) / TPB, TPB, 0, stream>>>(p1, ux1, zx1, 4 * C_B * C_INPUT);
  mask_kernel<<<(4 * GH + TPB - 1) / TPB, TPB, 0, stream>>>(p1, uh1, zh1, 4 * GH);
  mask_kernel<<<(4 * GH + TPB - 1) / TPB, TPB, 0, stream>>>(p2, ux2, zx2, 4 * GH);
  mask_kernel<<<(4 * GH + TPB - 1) / TPB, TPB, 0, stream>>>(p2, uh2, zh2, 4 * GH);
  mask_kernel<<<(GH + TPB - 1) / TPB, TPB, 0, stream>>>(p_out, u_out, zo, GH);

  wcat_kernel<<<4096, TPB, 0, stream>>>(Wx1, Wh1, Wcat1, C_INPUT, C_HID);
  wcat_kernel<<<4096, TPB, 0, stream>>>(Wx2, Wh2, Wcat2, C_HID, C_HID);

  zero_kernel<<<1024, TPB, 0, stream>>>((unsigned int*)A1, 4L * C_B * 1088 * 2 / 4);
  zero_kernel<<<1024, TPB, 0, stream>>>((unsigned int*)A2, 4L * C_B * 2048 * 2 / 4);
  zero_kernel<<<1024, TPB, 0, stream>>>((unsigned int*)c1, GH);
  zero_kernel<<<1024, TPB, 0, stream>>>((unsigned int*)c2, GH);

  // -------- interleaved recurrent scan (layer1 step t, then layer2 step t) ----
  // grid: (N/64, M/32, gates); block: 2 waves, each 32x32 register tile.
  dim3 ggrid(C_HID / 64, C_B / 32, 4);
  for (int t = 0; t < C_T; ++t) {
    prep1_kernel<<<(4 * C_B * C_INPUT + TPB - 1) / TPB, TPB, 0, stream>>>(x, zx1, A1, t);
    gemm_bf16_kernel<<<ggrid, 64, 0, stream>>>((const __bf16*)A1, (const __bf16*)Wcat1, pre1, 1088);
    elem1_kernel<<<(GH + TPB - 1) / TPB, TPB, 0, stream>>>(pre1, bx1, bh1, zh1, zx2, c1, A1, A2);
    gemm_bf16_kernel<<<ggrid, 64, 0, stream>>>((const __bf16*)A2, (const __bf16*)Wcat2, pre2, 2048);
    elem2_kernel<<<(GH + TPB - 1) / TPB, TPB, 0, stream>>>(pre2, bx2, bh2, zh2, c2, A2, h2last);
  }

  // -------- batchnorm (only t=127 feeds the head) + masked linear head ------
  stats_kernel<<<1, 1024, 0, stream>>>(h2last, stats);
  out_kernel<<<C_B, 256, 0, stream>>>(h2last, stats, zo, Wmu, bmu, gamma, beta, out);
}
